// MYLSTMWithGateoutput_78829829750962
// MI455X (gfx1250) — matrix operand order repaired and checked by reading, not compiled, not run on hardware
//
#include <hip/hip_runtime.h>
#include <hip/hip_bf16.h>

// Problem dims (from reference): B=8, T=512, D=2048, H=2048
#define DD      2048
#define HHDIM   2048
#define FOURH   8192
#define STEPS   4096        // B*T flattened sequential steps
#define NWG     128         // persistent scan workgroups

typedef unsigned short u16;
typedef unsigned int   u32;
typedef __attribute__((ext_vector_type(8)))  u16    u16x8;
typedef __attribute__((ext_vector_type(16))) __bf16 v16bf;
typedef __attribute__((ext_vector_type(8)))  float  v8f;

// builtin's parameter type: vector_size(16) int, AS(1) src / AS(3) dst
typedef int v4i __attribute__((vector_size(16)));
typedef __attribute__((address_space(1))) v4i gv4i;
typedef __attribute__((address_space(3))) v4i lv4i;

// ---------------- workspace layout (bytes) ----------------
#define XP_OFF   ((size_t)0)                          // 4096*8192 f32 = 128 MiB
#define XBF_OFF  (XP_OFF  + (size_t)STEPS*FOURH*4)    // 4096*2048 bf16
#define WIH_OFF  (XBF_OFF + (size_t)STEPS*DD*2)       // 8192*2048 bf16
#define WHH_OFF  (WIH_OFF + (size_t)FOURH*DD*2)       // 8192*2048 bf16
#define H_OFF    (WHH_OFF + (size_t)FOURH*HHDIM*2)
#define C_OFF    (H_OFF   + (size_t)HHDIM*4)
#define PRE_OFF  (C_OFF   + (size_t)HHDIM*4)
#define BAR_OFF  (PRE_OFF + (size_t)FOURH*4)

__device__ __forceinline__ u16 f32_to_bf16_rne(float f) {
  u32 u = __float_as_uint(f);
  u32 r = 0x7FFFu + ((u >> 16) & 1u);
  return (u16)((u + r) >> 16);
}
__device__ __forceinline__ float bf16_to_f32(u16 v) {
  return __uint_as_float(((u32)v) << 16);
}

union FragU { u16x8 h[2]; v16bf v; };

// ---------------- CDNA5 async global->LDS path (guarded) ----------------
#if __has_builtin(__builtin_amdgcn_global_load_async_to_lds_b128)
#define HAVE_ASYNC_LDS 1
__device__ __forceinline__ void async_cp16(const u16* g, u16* l) {
  __builtin_amdgcn_global_load_async_to_lds_b128(
      (gv4i*)(void*)const_cast<u16*>(g),
      (lv4i*)(void*)l, 0, 0);
}
#else
#define HAVE_ASYNC_LDS 0
#endif

__device__ __forceinline__ void wait_async_le4() {
#if HAVE_ASYNC_LDS
#if __has_builtin(__builtin_amdgcn_s_wait_asynccnt)
  __builtin_amdgcn_s_wait_asynccnt(4);
#else
  asm volatile("s_wait_asynccnt 4" ::: "memory");
#endif
#endif
}
__device__ __forceinline__ void wait_async_0() {
#if HAVE_ASYNC_LDS
#if __has_builtin(__builtin_amdgcn_s_wait_asynccnt)
  __builtin_amdgcn_s_wait_asynccnt(0);
#else
  asm volatile("s_wait_asynccnt 0" ::: "memory");
#endif
#endif
}

// ---------------- prep: fp32 -> bf16 ----------------
__global__ void lstm_cvt_bf16(const float* __restrict__ s, u16* __restrict__ d, int n) {
  int i = blockIdx.x * blockDim.x + threadIdx.x;
  int stride = gridDim.x * blockDim.x;
  for (; i < n; i += stride) d[i] = f32_to_bf16_rne(s[i]);
}

// ---------------- prep: seed h, c, barrier ----------------
__global__ void lstm_init_state(const float* __restrict__ h0, const float* __restrict__ c0,
                                float* __restrict__ h, float* __restrict__ c,
                                u32* __restrict__ bar) {
  int i = blockIdx.x * blockDim.x + threadIdx.x;
  if (i < HHDIM) { h[i] = h0[i]; c[i] = c0[i]; }
  if (i < 2) bar[i] = 0u;
}

// ---------------- WMMA GEMM: xp[m][n] = sum_k x[m][k]*Wih[n][k] + bih[n] ----------------
// 128x128 tile per WG; 8 waves, each 32(M)x64(N) = 2x4 WMMA accumulators; K-step 32.
// Double-buffered LDS tiles filled by GLOBAL_LOAD_ASYNC_TO_LDS_B128 (ASYNCcnt pipelined).
__global__ __launch_bounds__(256)
void lstm_wmma_xproj(const u16* __restrict__ xbf, const u16* __restrict__ wbf,
                     const float* __restrict__ bias, float* __restrict__ xp) {
  __shared__ u16 As[2][128 * 32];  // [buf][m][k]
  __shared__ u16 Bs[2][128 * 32];  // [buf][n][k]

  const int nBase = blockIdx.x * 128;
  const int mBase = blockIdx.y * 128;
  const int tid   = threadIdx.x;
  const int wave  = tid >> 5;
  const int lane  = tid & 31;
  const int waveM = wave & 3;     // 4 M-slots of 32 rows
  const int waveN = wave >> 2;    // 2 N-slots of 64 cols
  const int g     = lane >> 4;    // half-wave group
  const int ml    = lane & 15;

  v8f acc[2][4];
#pragma unroll
  for (int a = 0; a < 2; ++a)
#pragma unroll
    for (int b = 0; b < 4; ++b)
      acc[a][b] = (v8f){0.f, 0.f, 0.f, 0.f, 0.f, 0.f, 0.f, 0.f};

  const int mLd = tid >> 1;            // row handled by this thread
  const int kLd = (tid & 1) * 16;      // k-chunk within 32

  // issue one 128x32 A panel + 128x32 B panel fill (4 x 16B per thread)
  auto issue_tile = [&](int k0, int buf) {
    const u16* ga = xbf + (size_t)(mBase + mLd) * DD + k0 + kLd;
    const u16* gb = wbf + (size_t)(nBase + mLd) * DD + k0 + kLd;
    u16* la = &As[buf][mLd * 32 + kLd];
    u16* lb = &Bs[buf][mLd * 32 + kLd];
#if HAVE_ASYNC_LDS
    async_cp16(ga,     la);
    async_cp16(ga + 8, la + 8);
    async_cp16(gb,     lb);
    async_cp16(gb + 8, lb + 8);
#else
    *(u16x8*)la       = *(const u16x8*)ga;
    *(u16x8*)(la + 8) = *(const u16x8*)(ga + 8);
    *(u16x8*)lb       = *(const u16x8*)gb;
    *(u16x8*)(lb + 8) = *(const u16x8*)(gb + 8);
#endif
  };

  issue_tile(0, 0);
  int kb = 0;
  for (int k0 = 0; k0 < DD; k0 += 32, kb ^= 1) {
    if (k0 + 32 < DD) {
      issue_tile(k0 + 32, kb ^ 1);   // prefetch next tile into other buffer
      wait_async_le4();              // current tile's 4 async ops have landed
    } else {
      wait_async_0();
    }
    __syncthreads();                 // all waves' fills of buf[kb] visible

    // A fragments: 16-bit A 16x32 layout -> two 16B chunks at k = g*8 and 16+g*8
    v16bf afrag[2];
#pragma unroll
    for (int ms = 0; ms < 2; ++ms) {
      int m = waveM * 32 + ms * 16 + ml;
      FragU f;
      f.h[0] = *(const u16x8*)&As[kb][m * 32 + g * 8];
      f.h[1] = *(const u16x8*)&As[kb][m * 32 + 16 + g * 8];
      afrag[ms] = f.v;
    }
    // B fragments: 16-bit B 32x16 layout -> two 16B chunks at k = g*8 and 16+g*8 for column n
    v16bf bfrag[4];
#pragma unroll
    for (int ns = 0; ns < 4; ++ns) {
      int n = waveN * 64 + ns * 16 + ml;
      FragU f;
      f.h[0] = *(const u16x8*)&Bs[kb][n * 32 + g * 8];
      f.h[1] = *(const u16x8*)&Bs[kb][n * 32 + 16 + g * 8];
      bfrag[ns] = f.v;
    }
#pragma unroll
    for (int ms = 0; ms < 2; ++ms)
#pragma unroll
      for (int ns = 0; ns < 4; ++ns)
        acc[ms][ns] = __builtin_amdgcn_wmma_f32_16x16x32_bf16(
            false, afrag[ms], false, bfrag[ns], (short)0, acc[ms][ns], false, false);

    __syncthreads();                 // buf[kb] fully consumed before it is refilled
  }

  // epilogue: C/D layout -> element (M = g*8 + r, N = lane%16) in VGPR r
#pragma unroll
  for (int ns = 0; ns < 4; ++ns) {
    int col = nBase + waveN * 64 + ns * 16 + ml;
    float bv = bias[col];
#pragma unroll
    for (int ms = 0; ms < 2; ++ms) {
#pragma unroll
      for (int r = 0; r < 8; ++r) {
        int row = mBase + waveM * 32 + ms * 16 + g * 8 + r;
        xp[(size_t)row * FOURH + col] = acc[ms][ns][r] + bv;
      }
    }
  }
}

// ---------------- device-wide barrier (monotone generation) ----------------
__device__ __forceinline__ void lstm_grid_sync(u32* count, u32* gen, u32 target) {
  __threadfence();
  __syncthreads();
  if (threadIdx.x == 0) {
    u32 prev = __hip_atomic_fetch_add(count, 1u, __ATOMIC_ACQ_REL, __HIP_MEMORY_SCOPE_AGENT);
    if (prev == target * NWG - 1u) {
      __hip_atomic_store(gen, target, __ATOMIC_RELEASE, __HIP_MEMORY_SCOPE_AGENT);
    } else {
      while (__hip_atomic_load(gen, __ATOMIC_ACQUIRE, __HIP_MEMORY_SCOPE_AGENT) < target)
        __builtin_amdgcn_s_sleep(2);
    }
  }
  __syncthreads();
  __threadfence();
}

// ---------------- persistent sequential LSTM scan ----------------
// W_hh held bf16 (32 MB -> L2-resident on MI455X's 192 MB L2).
__global__ __launch_bounds__(256)
void lstm_scan(const float* __restrict__ xp, const u16* __restrict__ whh,
               const float* __restrict__ bhh,
               float* __restrict__ h, float* __restrict__ c, float* __restrict__ pre,
               float* __restrict__ out_h, float* __restrict__ out_f, float* __restrict__ out_i,
               u32* __restrict__ bar) {
  __shared__ float h_lds[HHDIM];
  const int wg   = blockIdx.x;
  const int tid  = threadIdx.x;
  const int wave = tid >> 5;
  const int lane = tid & 31;
  u32* bar_count = bar;
  u32* bar_gen   = bar + 1;
  u32 target = 0;

  for (int t = 0; t < STEPS; ++t) {
    // stage h into LDS (current h is globally visible: kernel-entry or post-barrier)
    __syncthreads();
    for (int j = tid; j < HHDIM; j += 256) h_lds[j] = h[j];
    __syncthreads();

    // matvec: 64 rows per WG, 8 rows per wave; lanes stride packed-bf16 pairs
    const float* xprow = xp + (size_t)t * FOURH;
    for (int rr = 0; rr < 8; ++rr) {
      int row = wg * 64 + wave * 8 + rr;
      const u32* wrow = (const u32*)(whh + (size_t)row * HHDIM);
      float sum = 0.f;
#pragma unroll 4
      for (int i = 0; i < HHDIM / 64; ++i) {
        int p = lane + i * 32;           // pair index: covers k = 2p, 2p+1
        u32 pk = wrow[p];
        sum = fmaf(bf16_to_f32((u16)(pk & 0xFFFFu)), h_lds[2 * p],     sum);
        sum = fmaf(bf16_to_f32((u16)(pk >> 16)),     h_lds[2 * p + 1], sum);
      }
#pragma unroll
      for (int off = 16; off > 0; off >>= 1) sum += __shfl_down(sum, off, 32);
      if (lane == 0) pre[row] = sum + xprow[row] + bhh[row];
    }
    lstm_grid_sync(bar_count, bar_gen, ++target);   // pre[] complete device-wide

    // gates: 16 H-elements per WG
    if (tid < 16) {
      int idx = wg * 16 + tid;
      float pi = pre[idx];
      float pf = pre[idx + HHDIM];
      float po = pre[idx + 2 * HHDIM];
      float pg = pre[idx + 3 * HHDIM];
      float it = 1.f / (1.f + __expf(-pi));
      float ft = 1.f / (1.f + __expf(-pf));
      float ot = 1.f / (1.f + __expf(-po));
      float gt = tanhf(pg);
      float cn = c[idx] * ft + it * gt;
      float hn = ot * tanhf(cn);
      c[idx] = cn;
      h[idx] = hn;
      size_t o = (size_t)t * HHDIM + idx;
      out_h[o] = hn;
      out_f[o] = ft;
      out_i[o] = it;
    }
    lstm_grid_sync(bar_count, bar_gen, ++target);   // h/c updated device-wide
  }
}

extern "C" void kernel_launch(void* const* d_in, const int* in_sizes, int n_in,
                              void* d_out, int out_size, void* d_ws, size_t ws_size,
                              hipStream_t stream) {
  const float* x   = (const float*)d_in[0];   // [4096, 2048]
  const float* h0  = (const float*)d_in[1];   // [2048]
  const float* c0  = (const float*)d_in[2];   // [2048]
  const float* Wih = (const float*)d_in[3];   // [8192, 2048]
  const float* bih = (const float*)d_in[4];   // [8192]
  const float* Whh = (const float*)d_in[5];   // [8192, 2048]
  const float* bhh = (const float*)d_in[6];   // [8192]
  float* out = (float*)d_out;

  char* ws = (char*)d_ws;
  float* xp    = (float*)(ws + XP_OFF);
  u16*   xbf   = (u16*)  (ws + XBF_OFF);
  u16*   wihbf = (u16*)  (ws + WIH_OFF);
  u16*   whhbf = (u16*)  (ws + WHH_OFF);
  float* hbuf  = (float*)(ws + H_OFF);
  float* cbuf  = (float*)(ws + C_OFF);
  float* pre   = (float*)(ws + PRE_OFF);
  u32*   bar   = (u32*)  (ws + BAR_OFF);

  // precision prep (bf16 multiplicands for WMMA / L2-resident W_hh)
  lstm_cvt_bf16<<<4096, 256, 0, stream>>>(x,   xbf,   STEPS * DD);
  lstm_cvt_bf16<<<4096, 256, 0, stream>>>(Wih, wihbf, FOURH * DD);
  lstm_cvt_bf16<<<4096, 256, 0, stream>>>(Whh, whhbf, FOURH * HHDIM);
  lstm_init_state<<<8, 256, 0, stream>>>(h0, c0, hbuf, cbuf, bar);

  // big parallel GEMM: xp = x @ Wih^T + bih  (v_wmma_f32_16x16x32_bf16)
  lstm_wmma_xproj<<<dim3(FOURH / 128, STEPS / 128), 256, 0, stream>>>(xbf, wihbf, bih, xp);

  // persistent sequential scan with device-wide barriers
  float* out_h = out;
  float* out_f = out + (size_t)STEPS * HHDIM;
  float* out_i = out + (size_t)2 * STEPS * HHDIM;
  lstm_scan<<<NWG, 256, 0, stream>>>(xp, whhbf, bhh, hbuf, cbuf, pre,
                                     out_h, out_f, out_i, bar);
}
